// RNN_61761629716481
// MI455X (gfx1250) — matrix operand order repaired and checked by reading, not compiled, not run on hardware
//
#include <hip/hip_runtime.h>
#include <hip/hip_bf16.h>
#include <math.h>

#define B_ 64
#define T_ 512
#define I_ 1024
#define H_ 1024
#define KDIM 1024            // == I_ == H_ (contraction dim for both GEMMs)
#define LDS_PITCH 520        // 512 staged K elems + 8 pad (4-bank row stride)

typedef __attribute__((ext_vector_type(16))) __bf16         v16bf;
typedef __attribute__((ext_vector_type(8)))  float          v8f;
typedef __attribute__((ext_vector_type(8)))  unsigned short v8us;

union Frag {
    struct { v8us lo, hi; } h;
    v16bf v;
};

// Build a 16-element bf16 WMMA fragment from two 16-byte chunks.
__device__ inline v16bf load_frag(const unsigned short* p0, const unsigned short* p1) {
    Frag f;
    f.h.lo = *(const v8us*)p0;
    f.h.hi = *(const v8us*)p1;
    return f.v;
}

#define WMMA(a, b, c) \
    __builtin_amdgcn_wmma_f32_16x16x32_bf16(false, (a), false, (b), (short)0, (c), false, false)

// fp32 -> bf16 round-to-nearest-even
__device__ inline unsigned short f2bf(float x) {
    unsigned int u = __float_as_uint(x);
    u += 0x7FFFu + ((u >> 16) & 1u);
    return (unsigned short)(u >> 16);
}

// ---------------------------------------------------------------------------
// Phase 1:  C[M,N] = A[M,K] @ Bpacked + bias(N)     (xproj = x@Wx + bx + bh)
// Block = 8 waves, computes a 32-row x 512-col macro tile.
// A tile staged in LDS (two 512-wide K halves); each wave: 2 M-tiles x 4
// N-tiles = 8 f32 accumulators, so every B fragment feeds two WMMAs.
// ---------------------------------------------------------------------------
__global__ void __launch_bounds__(256)
wmma_gemm_bias(const unsigned short* __restrict__ A,
               const unsigned short* __restrict__ Bp,
               const float* __restrict__ bsum,
               float* __restrict__ C,
               int M, int N) {
    __shared__ unsigned short As[32 * LDS_PITCH];   // 33,280 B

    const int tid  = threadIdx.x;
    const int lane = tid & 31;
    const int w    = tid >> 5;            // wave in block: 0..7
    const int mt2  = blockIdx.x >> 1;     // 32-row macro tile index
    const int half = blockIdx.x & 1;      // which 512-col half of N

    const int ntg   = half * 8 + w;       // 64-col group: 0..15
    const int kBase = (lane >> 4) * 16;   // A-fragment K sub-offset (contiguous, matches packed B)
    const unsigned short* Atile = A + (size_t)mt2 * 32 * KDIM;
    const unsigned short* Bbase = Bp + (size_t)(ntg * 4) * (KDIM >> 5) * 512;
    const size_t tileStride = (size_t)(KDIM >> 5) * 512;  // packed elems per nt

    const unsigned short* a0row = &As[(lane & 15) * LDS_PITCH];
    const unsigned short* a1row = a0row + 16 * LDS_PITCH;

    v8f c00 = {}, c01 = {}, c02 = {}, c03 = {};
    v8f c10 = {}, c11 = {}, c12 = {}, c13 = {};

    for (int kh = 0; kh < 2; ++kh) {
        __syncthreads();   // previous half's LDS reads complete
        // ---- cooperative stage: 32 rows x 512 K elems (16-B chunks) ----
        const unsigned short* Asrc = Atile + kh * 512;
        for (int c = tid; c < 32 * 64; c += 256) {
            int r  = c >> 6;       // row 0..31
            int kc = c & 63;       // 16-B chunk within row
            *(v8us*)&As[r * LDS_PITCH + kc * 8] =
                *(const v8us*)&Asrc[(size_t)r * KDIM + kc * 8];
        }
        __syncthreads();

        for (int kt = 0; kt < 16; ++kt) {
            const unsigned short* ap0 = a0row + kt * 32 + kBase;
            const unsigned short* ap1 = a1row + kt * 32 + kBase;
            v16bf a0 = load_frag(ap0, ap0 + 8);
            v16bf a1 = load_frag(ap1, ap1 + 8);

            const unsigned short* bp =
                Bbase + (size_t)(kh * 16 + kt) * 512 + lane * 16;
            if (kh * 16 + kt + 1 < 32)
                __builtin_prefetch((const void*)(bp + 512), 0, 3);

            v16bf b0 = load_frag(bp,                  bp + 8);
            v16bf b1 = load_frag(bp + tileStride,     bp + tileStride + 8);
            v16bf b2 = load_frag(bp + 2 * tileStride, bp + 2 * tileStride + 8);
            v16bf b3 = load_frag(bp + 3 * tileStride, bp + 3 * tileStride + 8);

            c00 = WMMA(a0, b0, c00);  c10 = WMMA(a1, b0, c10);
            c01 = WMMA(a0, b1, c01);  c11 = WMMA(a1, b1, c11);
            c02 = WMMA(a0, b2, c02);  c12 = WMMA(a1, b2, c12);
            c03 = WMMA(a0, b3, c03);  c13 = WMMA(a1, b3, c13);
        }
    }

    // C layout: vgpr r, lanes 0-15 -> (M=r, N=lane), lanes 16-31 -> (M=8+r)
    int colBase = ntg * 64 + (lane & 15);
    int rBase   = (lane >> 4) * 8;
    float bs0 = bsum[colBase];
    float bs1 = bsum[colBase + 16];
    float bs2 = bsum[colBase + 32];
    float bs3 = bsum[colBase + 48];
#pragma unroll
    for (int r = 0; r < 8; ++r) {
        size_t row0 = (size_t)(mt2 * 32 + rBase + r) * N;
        size_t row1 = row0 + (size_t)16 * N;
        C[row0 + colBase]      = c00[r] + bs0;
        C[row0 + colBase + 16] = c01[r] + bs1;
        C[row0 + colBase + 32] = c02[r] + bs2;
        C[row0 + colBase + 48] = c03[r] + bs3;
        C[row1 + colBase]      = c10[r] + bs0;
        C[row1 + colBase + 16] = c11[r] + bs1;
        C[row1 + colBase + 32] = c12[r] + bs2;
        C[row1 + colBase + 48] = c13[r] + bs3;
    }
}

// ---------------------------------------------------------------------------
// Recurrent step:  h_new = tanh(xproj_t + h_prev @ Wh)
// xproj_t lives in out[:, t, :] and is overwritten in place with h_new.
// Latency-bound (64 waves); direct-global loads, no LDS barriers on the
// serial critical path. Wh packed fragments stay hot in L2.
// ---------------------------------------------------------------------------
__global__ void __launch_bounds__(256)
wmma_step(const unsigned short* __restrict__ Hprev,
          const unsigned short* __restrict__ WhP,
          float* __restrict__ out,
          unsigned short* __restrict__ Hnext,
          int t) {
    const int NTG = H_ >> 6;  // 16
    int wave = (int)((blockIdx.x * blockDim.x + threadIdx.x) >> 5);
    int lane = threadIdx.x & 31;
    int mt  = wave / NTG;     // 0..3  (batch row tiles)
    int ntg = wave - mt * NTG;

    int row   = mt * 16 + (lane & 15);
    int kBase = (lane >> 4) * 16;
    const unsigned short* Arow  = Hprev + (size_t)row * H_;
    const unsigned short* Bbase = WhP + (size_t)(ntg * 4) * (H_ >> 5) * 512;
    const size_t tileStride = (size_t)(H_ >> 5) * 512;

    v8f c0 = {}, c1 = {}, c2 = {}, c3 = {};
    for (int kt = 0; kt < H_ / 32; ++kt) {
        const unsigned short* ap = Arow + kt * 32 + kBase;
        v16bf a = load_frag(ap, ap + 8);

        const unsigned short* bp = Bbase + (size_t)kt * 512 + lane * 16;
        if (kt + 1 < H_ / 32) __builtin_prefetch((const void*)(bp + 512), 0, 3);

        v16bf b0 = load_frag(bp,                  bp + 8);
        v16bf b1 = load_frag(bp + tileStride,     bp + tileStride + 8);
        v16bf b2 = load_frag(bp + 2 * tileStride, bp + 2 * tileStride + 8);
        v16bf b3 = load_frag(bp + 3 * tileStride, bp + 3 * tileStride + 8);

        c0 = WMMA(a, b0, c0);
        c1 = WMMA(a, b1, c1);
        c2 = WMMA(a, b2, c2);
        c3 = WMMA(a, b3, c3);
    }

    int colBase = ntg * 64 + (lane & 15);
    int rBase   = (lane >> 4) * 8;
#pragma unroll
    for (int r = 0; r < 8; ++r) {
        int b = mt * 16 + rBase + r;
        size_t base = ((size_t)b * T_ + t) * H_;
#pragma unroll
        for (int i = 0; i < 4; ++i) {
            int col = colBase + i * 16;
            float acc = (i == 0) ? c0[r] : (i == 1) ? c1[r] : (i == 2) ? c2[r] : c3[r];
            float v = tanhf(out[base + col] + acc);
            out[base + col] = v;
            Hnext[(size_t)b * H_ + col] = f2bf(v);
        }
    }
}

// ---------------------------------------------------------------------------
// Prep / epilogue helpers
// ---------------------------------------------------------------------------
__global__ void f32_to_bf16_kernel(const float* __restrict__ in,
                                   unsigned short* __restrict__ out, size_t n) {
    size_t i = (size_t)blockIdx.x * blockDim.x + threadIdx.x;
    if (i < n) out[i] = f2bf(in[i]);
}

// Repack W[K,N] (f32 row-major) into WMMA B-fragment order:
// tile (kt,nt) of 32x16 at ((nt*KT)+kt)*512; lane l holds column n=l&15,
// K range kt*32 + (l>>4)*16 .. +15, stored contiguously (32 B per lane).
__global__ void pack_weight_kernel(const float* __restrict__ W,
                                   unsigned short* __restrict__ P, int K, int N) {
    size_t p = (size_t)blockIdx.x * blockDim.x + threadIdx.x;
    if (p >= (size_t)K * N) return;
    int KT = K >> 5;
    size_t tl = p >> 9;         // /512
    int within = (int)(p & 511);
    int lane = within >> 4;
    int j = within & 15;
    int nt = (int)(tl / KT);
    int kt = (int)(tl - (size_t)nt * KT);
    int n = nt * 16 + (lane & 15);
    int k = kt * 32 + (lane >> 4) * 16 + j;
    P[p] = f2bf(W[(size_t)k * N + n]);
}

__global__ void bias_sum_kernel(const float* __restrict__ bx,
                                const float* __restrict__ bh,
                                float* __restrict__ bs, int n) {
    int i = blockIdx.x * blockDim.x + threadIdx.x;
    if (i < n) bs[i] = bx[i] + bh[i];
}

__global__ void zero_u16_kernel(unsigned short* __restrict__ p, size_t n) {
    size_t i = (size_t)blockIdx.x * blockDim.x + threadIdx.x;
    if (i < n) p[i] = 0;
}

__global__ void copy_last_kernel(const float* __restrict__ outs,
                                 float* __restrict__ hlast) {
    int i = blockIdx.x * blockDim.x + threadIdx.x;
    if (i >= B_ * H_) return;
    int b = i >> 10;            // /H_
    int n = i & (H_ - 1);
    hlast[i] = outs[((size_t)b * T_ + (T_ - 1)) * H_ + n];
}

// ---------------------------------------------------------------------------
extern "C" void kernel_launch(void* const* d_in, const int* in_sizes, int n_in,
                              void* d_out, int out_size, void* d_ws, size_t ws_size,
                              hipStream_t stream) {
    const float* x  = (const float*)d_in[0];   // [B,T,I]
    const float* Wx = (const float*)d_in[1];   // [I,H]
    const float* bx = (const float*)d_in[2];   // [H]
    const float* Wh = (const float*)d_in[3];   // [H,H]
    const float* bh = (const float*)d_in[4];   // [H]
    float* out = (float*)d_out;                // outs [B,T,H] ++ h_last [B,H]

    // Workspace carve-up (all 2-byte counts are multiples of 4 bytes)
    unsigned short* xbf = (unsigned short*)d_ws;           // B*T*I bf16 (64 MB)
    unsigned short* WxP = xbf + (size_t)B_ * T_ * I_;      // I*H bf16   (2 MB)
    unsigned short* WhP = WxP + (size_t)I_ * H_;           // H*H bf16   (2 MB)
    unsigned short* h0  = WhP + (size_t)H_ * H_;           // B*H bf16
    unsigned short* h1  = h0 + (size_t)B_ * H_;            // B*H bf16
    float* bs = (float*)(h1 + (size_t)B_ * H_);            // H f32

    size_t nx = (size_t)B_ * T_ * I_;
    f32_to_bf16_kernel<<<(unsigned)((nx + 255) / 256), 256, 0, stream>>>(x, xbf, nx);
    pack_weight_kernel<<<(unsigned)(((size_t)I_ * H_ + 255) / 256), 256, 0, stream>>>(Wx, WxP, I_, H_);
    pack_weight_kernel<<<(unsigned)(((size_t)H_ * H_ + 255) / 256), 256, 0, stream>>>(Wh, WhP, H_, H_);
    bias_sum_kernel<<<(H_ + 255) / 256, 256, 0, stream>>>(bx, bh, bs, H_);
    zero_u16_kernel<<<(unsigned)(((size_t)B_ * H_ + 255) / 256), 256, 0, stream>>>(h0, (size_t)B_ * H_);

    // Phase 1: xproj for all timesteps, written into the outs slab in place.
    // Block covers 32 rows x 512 cols -> grid = (M/32) * (N/512).
    const int M = B_ * T_;                       // 32768
    wmma_gemm_bias<<<(M / 32) * (H_ / 512), 256, 0, stream>>>(xbf, WxP, bs, out, M, H_);

    // Phase 2: 512 dependent steps, h ping-pong between h0/h1.
    unsigned short* hb[2] = {h0, h1};
    for (int t = 0; t < T_; ++t) {
        wmma_step<<<8, 256, 0, stream>>>(hb[t & 1], WhP, out, hb[(t + 1) & 1], t);
    }

    // h_last = outs[:, T-1, :]
    copy_last_kernel<<<(B_ * H_ + 255) / 256, 256, 0, stream>>>(out, out + (size_t)B_ * T_ * H_);
}
